// VAE_7782480740489
// MI455X (gfx1250) — matrix operand order repaired and checked by reading, not compiled, not run on hardware
//
#include <hip/hip_runtime.h>

// ---------------------------------------------------------------------------
// Types
// ---------------------------------------------------------------------------
typedef __attribute__((ext_vector_type(16))) __bf16        v16bf;
typedef __attribute__((ext_vector_type(8)))  float         v8f;
typedef __attribute__((ext_vector_type(4)))  float         f32x4;
typedef __attribute__((ext_vector_type(4)))  unsigned int  u32x4;

union ABFrag {
    v16bf  v;
    __bf16 b[16];
    u32x4  q[2];
};

__device__ __forceinline__ float fast_tanh(float x) {
    x = fminf(fmaxf(x, -15.f), 15.f);
    float e = __expf(2.f * x);
    return (e - 1.f) * __builtin_amdgcn_rcpf(e + 1.f);
}

__device__ __forceinline__ v8f wmma_bf16(v16bf a, v16bf b, v8f c) {
    // D = A(16x32 bf16) x B(32x16 bf16) + C(16x16 f32)
    return __builtin_amdgcn_wmma_f32_16x16x32_bf16(
        /*neg_a=*/false, a, /*neg_b=*/false, b,
        /*c_mod=*/(short)0, c, /*reuse_a=*/false, /*reuse_b=*/false);
}

// ---------------------------------------------------------------------------
// Weight packing: W[N,K] (row major, torch Linear) -> bf16 WMMA B-fragment
// panels, zero padded.  Panel (nt,kt) is 32 lanes x 16 bf16 (1 KB):
//   lane l = n_local + 16*h holds W[nt*16+n_local, kt*32 + 8*h + j + (j>=8 ? 8 : 0)], j=0..15
// ---------------------------------------------------------------------------
__global__ void vae_pack_b_panels(const float* __restrict__ W, int N, int K,
                                  int nTiles, int kTiles,
                                  __bf16* __restrict__ dst) {
    int idx = blockIdx.x * blockDim.x + threadIdx.x;   // one per (panel, lane)
    int total = nTiles * kTiles * 32;
    if (idx >= total) return;
    int lane  = idx & 31;
    int panel = idx >> 5;
    int kt = panel % kTiles;
    int nt = panel / kTiles;
    int n  = nt * 16 + (lane & 15);
    int kb = kt * 32 + 8 * (lane >> 4);
    ABFrag f;
#pragma unroll
    for (int j = 0; j < 16; ++j) {
        int k = kb + j + (j >= 8 ? 8 : 0);
        float w = (n < N && k < K) ? W[(size_t)n * K + k] : 0.f;
        f.b[j] = (__bf16)w;
    }
    u32x4* o = (u32x4*)(dst + (size_t)idx * 16);
    o[0] = f.q[0];
    o[1] = f.q[1];
}

__global__ void vae_pack_bias(const float* __restrict__ b, int N, int Npad,
                              float* __restrict__ dst) {
    int i = blockIdx.x * blockDim.x + threadIdx.x;
    if (i < Npad) dst[i] = (i < N) ? b[i] : 0.f;
}

// ---------------------------------------------------------------------------
// Fragment loaders
// ---------------------------------------------------------------------------
__device__ __forceinline__ v16bf loadB(const __bf16* __restrict__ panel,
                                       int lane) {
    ABFrag f;
    const u32x4* p = (const u32x4*)(panel + (size_t)lane * 16);
    f.q[0] = p[0];
    f.q[1] = p[1];
    return f.v;
}

// A fragment from LDS staging buffer (row-major bf16, leading dim ldK).
// A layout: lane (m = lane&15, h = lane>>4): elems 0..7 = K0+8h+0..7,
// elems 8..15 = K0+8h+16..23.
__device__ __forceinline__ v16bf loadA_lds(const __bf16* st, int ldK,
                                           int kbase, int lane) {
    int m = lane & 15, h = lane >> 4;
    const __bf16* r = st + m * ldK + kbase + 8 * h;
    ABFrag f;
    f.q[0] = *(const u32x4*)(r);
    f.q[1] = *(const u32x4*)(r + 16);
    return f.v;
}

// A fragment straight from global f32 x (row stride 784), converting to bf16.
// K tail (>=784) chunks are zero-filled (tailOk==false iff kt==24).
__device__ __forceinline__ v16bf loadA_x(const float* __restrict__ xrow,
                                         int kt, int h, bool tailOk) {
    int k0 = kt * 32 + 8 * h;
    f32x4 c0 = *(const f32x4*)(xrow + k0);
    f32x4 c1 = *(const f32x4*)(xrow + k0 + 4);
    f32x4 c2 = {0.f, 0.f, 0.f, 0.f};
    f32x4 c3 = {0.f, 0.f, 0.f, 0.f};
    if (tailOk) {
        c2 = *(const f32x4*)(xrow + k0 + 16);
        c3 = *(const f32x4*)(xrow + k0 + 20);
    }
    ABFrag f;
#pragma unroll
    for (int i = 0; i < 4; ++i) {
        f.b[i]      = (__bf16)c0[i];
        f.b[4 + i]  = (__bf16)c1[i];
        f.b[8 + i]  = (__bf16)c2[i];
        f.b[12 + i] = (__bf16)c3[i];
    }
    return f.v;
}

__device__ __forceinline__ v8f bias8(const float* __restrict__ bp, int nt, int c) {
    float bv = bp[nt * 16 + c];
    v8f r;
#pragma unroll
    for (int i = 0; i < 8; ++i) r[i] = bv;
    return r;
}

// ---------------------------------------------------------------------------
// Fused VAE forward.  1 wave = 16 batch rows through the whole network.
// Padded dims: H_ENC 50->64, Z 25->32, H_DEC 100->112 (K-pad 128), X 784.
// ---------------------------------------------------------------------------
#define WPB 8  // waves per block

__global__ __launch_bounds__(256, 2)
void vae_fused(const float* __restrict__ x,
               const float* __restrict__ eps_z,
               const float* __restrict__ eps_x,
               const __bf16* __restrict__ w1,  // enc in   [4 x 25]
               const __bf16* __restrict__ w2,  // enc mu   [2 x 2]
               const __bf16* __restrict__ w3,  // enc ls   [2 x 2]
               const __bf16* __restrict__ w4,  // dec in   [7 x 1]
               const __bf16* __restrict__ w5,  // dec mu   [49 x 4]
               const __bf16* __restrict__ w6,  // dec ls   [49 x 4]
               const float* __restrict__ b1, const float* __restrict__ b2,
               const float* __restrict__ b3, const float* __restrict__ b4,
               const float* __restrict__ b5, const float* __restrict__ b6,
               float* __restrict__ out, int rows) {
    __shared__ __align__(16) __bf16 stage[WPB][16 * 128];

    const int lane = threadIdx.x & 31;
    const int wave = threadIdx.x >> 5;
    const int row0 = (blockIdx.x * WPB + wave) * 16;
    if (row0 + 16 > rows) return;  // wave-uniform: EXEC stays all-ones

    const int c = lane & 15;  // tile column (B/C/D layout)
    const int h = lane >> 4;
    __bf16* st = &stage[wave][0];

    // ---- Encoder layer 1: h = tanh(x @ W1^T + b1), K=784 (25 kt), N=64 (4 nt)
    const float* xrow = x + (size_t)(row0 + c) * 784;  // A-row for this lane
    v8f hE[4];
#pragma unroll
    for (int nt = 0; nt < 4; ++nt) hE[nt] = bias8(b1, nt, c);
    for (int kt = 0; kt < 25; ++kt) {
        v16bf a = loadA_x(xrow, kt, h, kt < 24);
#pragma unroll
        for (int nt = 0; nt < 4; ++nt) {
            v16bf b = loadB(w1 + ((size_t)(nt * 25 + kt) << 9), lane);
            hE[nt] = wmma_bf16(a, b, hE[nt]);
        }
    }
    // stage h as bf16 [16 x 64]
#pragma unroll
    for (int nt = 0; nt < 4; ++nt) {
#pragma unroll
        for (int r = 0; r < 8; ++r) {
            int m = r + 8 * h;
            st[m * 64 + nt * 16 + c] = (__bf16)fast_tanh(hE[nt][r]);
        }
    }
    asm volatile("s_wait_dscnt 0" ::: "memory");

    // ---- Encoder heads: mu_e, ls_e. K=64 (2 kt), N=32 (2 nt) each
    v16bf aH0 = loadA_lds(st, 64, 0, lane);
    v16bf aH1 = loadA_lds(st, 64, 32, lane);
    v8f mu[2], ls[2];
#pragma unroll
    for (int nt = 0; nt < 2; ++nt) {
        mu[nt] = bias8(b2, nt, c);
        ls[nt] = bias8(b3, nt, c);
    }
#pragma unroll
    for (int nt = 0; nt < 2; ++nt) {
        mu[nt] = wmma_bf16(aH0, loadB(w2 + ((size_t)(nt * 2 + 0) << 9), lane), mu[nt]);
        mu[nt] = wmma_bf16(aH1, loadB(w2 + ((size_t)(nt * 2 + 1) << 9), lane), mu[nt]);
        ls[nt] = wmma_bf16(aH0, loadB(w3 + ((size_t)(nt * 2 + 0) << 9), lane), ls[nt]);
        ls[nt] = wmma_bf16(aH1, loadB(w3 + ((size_t)(nt * 2 + 1) << 9), lane), ls[nt]);
    }
    // ---- Reparameterize: z = mu + exp(0.5 ls) * eps_z; stage [16 x 32] bf16
#pragma unroll
    for (int nt = 0; nt < 2; ++nt) {
        int col = nt * 16 + c;
#pragma unroll
        for (int r = 0; r < 8; ++r) {
            int m = r + 8 * h;
            float e = (col < 25) ? eps_z[(size_t)(row0 + m) * 25 + col] : 0.f;
            float zv = mu[nt][r] + __expf(0.5f * ls[nt][r]) * e;
            st[m * 32 + col] = (__bf16)zv;   // pad cols give exactly 0
        }
    }
    asm volatile("s_wait_dscnt 0" ::: "memory");

    // ---- Decoder layer 1: hd = tanh(z @ W4^T + b4). K=32 (1 kt), N=112 (7 nt)
    v16bf aZ = loadA_lds(st, 32, 0, lane);
    asm volatile("" ::: "memory");  // keep aZ loads before stage overwrites
    {   // zero K-pad tail cols 112..127 of the [16 x 128] hd staging
        u32x4 z4 = {0u, 0u, 0u, 0u};
        *(u32x4*)(st + c * 128 + 112 + 8 * h) = z4;
    }
#pragma unroll
    for (int nt = 0; nt < 7; ++nt) {
        v8f acc = bias8(b4, nt, c);
        acc = wmma_bf16(aZ, loadB(w4 + ((size_t)nt << 9), lane), acc);
#pragma unroll
        for (int r = 0; r < 8; ++r) {
            int m = r + 8 * h;
            st[m * 128 + nt * 16 + c] = (__bf16)fast_tanh(acc[r]);
        }
    }
    asm volatile("s_wait_dscnt 0" ::: "memory");

    // ---- Decoder heads: out = mu_d + exp(0.5 ls_d) * eps_x
    // K=128 (4 kt, tail zero), N=784 (49 nt).  Keep all A-frags in registers.
    v16bf aD[4];
#pragma unroll
    for (int kt = 0; kt < 4; ++kt) aD[kt] = loadA_lds(st, 128, kt * 32, lane);

    for (int nt = 0; nt < 49; ++nt) {
        v8f mud = bias8(b5, nt, c);
        v8f lsd = bias8(b6, nt, c);
#pragma unroll
        for (int kt = 0; kt < 4; ++kt) {
            mud = wmma_bf16(aD[kt], loadB(w5 + ((size_t)(nt * 4 + kt) << 9), lane), mud);
            lsd = wmma_bf16(aD[kt], loadB(w6 + ((size_t)(nt * 4 + kt) << 9), lane), lsd);
        }
        int col = nt * 16 + c;
#pragma unroll
        for (int r = 0; r < 8; ++r) {
            int m = r + 8 * h;
            size_t idx = (size_t)(row0 + m) * 784 + col;
            out[idx] = mud[r] + __expf(0.5f * lsd[r]) * eps_x[idx];
        }
    }
}

// ---------------------------------------------------------------------------
// Host launcher
// ---------------------------------------------------------------------------
extern "C" void kernel_launch(void* const* d_in, const int* in_sizes, int n_in,
                              void* d_out, int out_size, void* d_ws, size_t ws_size,
                              hipStream_t stream) {
    const float* x     = (const float*)d_in[0];
    const float* eps_z = (const float*)d_in[1];
    const float* eps_x = (const float*)d_in[2];
    const float* W1 = (const float*)d_in[3];  const float* B1 = (const float*)d_in[4];
    const float* W2 = (const float*)d_in[5];  const float* B2 = (const float*)d_in[6];
    const float* W3 = (const float*)d_in[7];  const float* B3 = (const float*)d_in[8];
    const float* W4 = (const float*)d_in[9];  const float* B4 = (const float*)d_in[10];
    const float* W5 = (const float*)d_in[11]; const float* B5 = (const float*)d_in[12];
    const float* W6 = (const float*)d_in[13]; const float* B6 = (const float*)d_in[14];

    const int rows = in_sizes[0] / 784;  // B = 131072

    // Workspace layout (bytes). Panels: nT*kT*512 bf16 each.
    char* ws = (char*)d_ws;
    __bf16* w1 = (__bf16*)(ws + 0);       // 4*25*512*2 = 102400
    __bf16* w2 = (__bf16*)(ws + 102400);  // 2*2*512*2  =   4096
    __bf16* w3 = (__bf16*)(ws + 106496);  //            =   4096
    __bf16* w4 = (__bf16*)(ws + 110592);  // 7*1*512*2  =   7168
    __bf16* w5 = (__bf16*)(ws + 117760);  // 49*4*512*2 = 200704
    __bf16* w6 = (__bf16*)(ws + 318464);  //            = 200704
    float* pb1 = (float*)(ws + 519168);  // 64
    float* pb2 = (float*)(ws + 519424);  // 32
    float* pb3 = (float*)(ws + 519552);  // 32
    float* pb4 = (float*)(ws + 519680);  // 112
    float* pb5 = (float*)(ws + 520128);  // 784
    float* pb6 = (float*)(ws + 523264);  // 784  (end = 526400 bytes)

    auto packW = [&](const float* W, int N, int K, int nT, int kT,
                     __bf16* dst) {
        int total = nT * kT * 32;
        hipLaunchKernelGGL(vae_pack_b_panels, dim3((total + 255) / 256),
                           dim3(256), 0, stream, W, N, K, nT, kT, dst);
    };
    packW(W1, 50, 784, 4, 25, w1);
    packW(W2, 25, 50, 2, 2, w2);
    packW(W3, 25, 50, 2, 2, w3);
    packW(W4, 100, 25, 7, 1, w4);
    packW(W5, 784, 100, 49, 4, w5);
    packW(W6, 784, 100, 49, 4, w6);

    auto packB = [&](const float* b, int N, int Np, float* dst) {
        hipLaunchKernelGGL(vae_pack_bias, dim3((Np + 255) / 256), dim3(256), 0,
                           stream, b, N, Np, dst);
    };
    packB(B1, 50, 64, pb1);
    packB(B2, 25, 32, pb2);
    packB(B3, 25, 32, pb3);
    packB(B4, 100, 112, pb4);
    packB(B5, 784, 784, pb5);
    packB(B6, 784, 784, pb6);

    const int blocks = rows / (WPB * 16);  // 131072 / 128 = 1024
    hipLaunchKernelGGL(vae_fused, dim3(blocks), dim3(256), 0, stream,
                       x, eps_z, eps_x, w1, w2, w3, w4, w5, w6,
                       pb1, pb2, pb3, pb4, pb5, pb6,
                       (float*)d_out, rows);
}
